// TransformerLMHead_46308337386164
// MI455X (gfx1250) — compile-verified, matrix operand order checked by reading, not run on hardware
//
#include <hip/hip_runtime.h>
#include <hip/hip_bf16.h>
#include <math.h>

// Problem dims (fixed by reference)
#define BB  2
#define SS  1024
#define DD  1024
#define DKK 64
#define DVV 64
#define HH  16
#define DHH 4096
#define LL  8
#define VV  256

typedef __attribute__((ext_vector_type(16))) __bf16 v16bf;
typedef __attribute__((ext_vector_type(8)))  float  v8f;

union Frag { v16bf v; unsigned short u[16]; };

// Native f32 -> bf16 (hardware cvt on gfx1250)
__device__ __forceinline__ unsigned short f2bf(float f) {
  union { __bf16 h; unsigned short u; } c;
  c.h = (__bf16)f;
  return c.u;
}
__device__ __forceinline__ unsigned int pk2bf(float x, float y) {
  return (unsigned int)f2bf(x) | ((unsigned int)f2bf(y) << 16);
}

// ---------------------------------------------------------------------------
// Embedding: x[b,s,:] = emb[tok[b,s],:] + pos[s,:]   (float4 vectorized)
// ---------------------------------------------------------------------------
__global__ __launch_bounds__(256) void embed_kernel(
    const int* __restrict__ tok, const float* __restrict__ emb,
    const float* __restrict__ pos, float* __restrict__ x)
{
  int i  = blockIdx.x * 256 + threadIdx.x;   // over B*S*(D/4)
  int dq = i % (DD / 4);
  int bs = i / (DD / 4);
  int s  = bs % SS;
  float4 e = ((const float4*)emb)[tok[bs] * (DD / 4) + dq];
  float4 p = ((const float4*)pos)[s * (DD / 4) + dq];
  float4 o; o.x = e.x + p.x; o.y = e.y + p.y; o.z = e.z + p.z; o.w = e.w + p.w;
  ((float4*)x)[i] = o;
}

// ---------------------------------------------------------------------------
// Weight conversion f32 -> bf16, pre-swizzled into WMMA B-fragment order:
//   flat = ((g*(K/32) + kb)*32 + lane)*16 + e
//   g = n>>4, kb = k>>5, lane = (n&15) + 16*((k>>3)&1), e = (k&7) + 8*((k&31)>>4)
// After this, a B fragment is 32 contiguous bytes per lane in global memory.
// ---------------------------------------------------------------------------
__global__ __launch_bounds__(256) void wconv_kernel(
    const float* __restrict__ W, unsigned short* __restrict__ Wp, int K, int N)
{
  int i  = blockIdx.x * 256 + threadIdx.x;   // over K*(N/8)
  int k  = i / (N / 8);
  int g8 = i % (N / 8);
  const float4* p = (const float4*)(W + (size_t)k * N + g8 * 8);
  float4 x0 = p[0], x1 = p[1];
  float v[8] = {x0.x, x0.y, x0.z, x0.w, x1.x, x1.y, x1.z, x1.w};
  const int kb = k >> 5;
  const int e  = (k & 7) + 8 * ((k & 31) >> 4);
  const int hk = (k >> 3) & 1;
  #pragma unroll
  for (int j = 0; j < 8; ++j) {
    int n = g8 * 8 + j;
    int g = n >> 4;
    int lane = (n & 15) + 16 * hk;
    Wp[((g * (K >> 5) + kb) * 32 + lane) * 16 + e] = f2bf(v[j]);
  }
}

// ---------------------------------------------------------------------------
// GEMM: C[M,N] = act(A[M,K] @ W[K,N] + bias) + res
// bf16 WMMA, f32 accumulate. Block 256 thr = 8 waves, tile 64x128, BK=32.
// W is bf16 pre-swizzled (B fragments = 2 global b128 per lane, straight from
// the L2-resident weight copy). A tile staged via 4KB LDS in fragment order.
// ABF16:   A is bf16 row-major (pure copy staging) vs f32 (convert staging).
// OUTMODE: 0 = f32 row-major, 1 = bf16 row-major, 2 = bf16 V-transposed
//          (Vt[b][h][dv][s], packed uint4 store of 8 consecutive tokens).
// ---------------------------------------------------------------------------
template<bool ABF16, bool BIAS, bool GELU_ACT, bool RES, int OUTMODE>
__global__ __launch_bounds__(256) void gemm_bf16_wmma(
    const void* __restrict__ Ain, const unsigned short* __restrict__ Wp,
    const float* __restrict__ bias, const float* __restrict__ res,
    void* __restrict__ Cout, int M, int N, int K)
{
  __shared__ __align__(16) unsigned int AsW[4 * 32 * 8];   // 4 row-bands (64 rows)

  const int tid  = threadIdx.x;
  const int wave = tid >> 5;
  const int lane = tid & 31;
  const int half = lane >> 4;
  const int lm   = lane & 15;
  const int rw   = wave & 1;    // row sub-band: 32 rows
  const int cw   = wave >> 1;   // col sub-band: 32 cols (2 groups of 16)
  const int mbase = blockIdx.y * 64;
  const int nbase = blockIdx.x * 128;
  const int kblocks = K >> 5;
  const int g0 = (nbase >> 4) + cw * 2;   // this wave's first 16-col group

  v8f acc[2][2] = {};

  // A staging: thread -> (row, k-octet); one uint4 LDS store
  const int arow = tid >> 2, aa = tid & 3;
  const int aband = arow >> 4;
  const int alane = (arow & 15) + 16 * (aa & 1);
  const int av0   = (aa >> 1) * 4;

  for (int k0 = 0; k0 < K; k0 += 32) {
    // ---- stage A tile (64x32) into fragment-ordered LDS ----
    uint4 apk;
    if (ABF16) {
      const unsigned short* Ab = (const unsigned short*)Ain;
      apk = *(const uint4*)(Ab + (size_t)(mbase + arow) * K + k0 + aa * 8);
    } else {
      const float* Af = (const float*)Ain;
      const float4* ap = (const float4*)(Af + (size_t)(mbase + arow) * K + k0 + aa * 8);
      float4 a0 = ap[0], a1 = ap[1];
      apk.x = pk2bf(a0.x, a0.y); apk.y = pk2bf(a0.z, a0.w);
      apk.z = pk2bf(a1.x, a1.y); apk.w = pk2bf(a1.z, a1.w);
    }
    *(uint4*)&AsW[(aband * 32 + alane) * 8 + av0] = apk;
    __syncthreads();

    // ---- fragments ----
    Frag a0f, a1f, b0f, b1f;
    {
      const uint4* p = (const uint4*)&AsW[((rw * 2) * 32 + lane) * 8];
      *(uint4*)&a0f.u[0] = p[0]; *(uint4*)&a0f.u[8] = p[1];
      p = (const uint4*)&AsW[((rw * 2 + 1) * 32 + lane) * 8];
      *(uint4*)&a1f.u[0] = p[0]; *(uint4*)&a1f.u[8] = p[1];
      // B fragments straight from swizzled global weights (L2-resident)
      const uint4* pb = (const uint4*)&Wp[(((g0)     * kblocks + (k0 >> 5)) * 32 + lane) * 16];
      *(uint4*)&b0f.u[0] = pb[0]; *(uint4*)&b0f.u[8] = pb[1];
      pb = (const uint4*)&Wp[(((g0 + 1) * kblocks + (k0 >> 5)) * 32 + lane) * 16];
      *(uint4*)&b1f.u[0] = pb[0]; *(uint4*)&b1f.u[8] = pb[1];
    }
    acc[0][0] = __builtin_amdgcn_wmma_f32_16x16x32_bf16(false, a0f.v, false, b0f.v,
                                                        (short)0, acc[0][0], false, false);
    acc[0][1] = __builtin_amdgcn_wmma_f32_16x16x32_bf16(false, a0f.v, false, b1f.v,
                                                        (short)0, acc[0][1], false, false);
    acc[1][0] = __builtin_amdgcn_wmma_f32_16x16x32_bf16(false, a1f.v, false, b0f.v,
                                                        (short)0, acc[1][0], false, false);
    acc[1][1] = __builtin_amdgcn_wmma_f32_16x16x32_bf16(false, a1f.v, false, b1f.v,
                                                        (short)0, acc[1][1], false, false);
    __syncthreads();
  }

  // Epilogue (C layout: col = lm, row = r + 8*half within each 16x16 tile)
  #pragma unroll
  for (int m = 0; m < 2; ++m) {
    #pragma unroll
    for (int n = 0; n < 2; ++n) {
      const int colc  = nbase + cw * 32 + n * 16 + lm;
      const int rbase = mbase + rw * 32 + m * 16 + 8 * half;
      float vals[8];
      #pragma unroll
      for (int r = 0; r < 8; ++r) {
        float x = acc[m][n][r];
        if (BIAS) x += bias[colc];
        if (GELU_ACT) x = 0.5f * x * (1.0f + erff(x * 0.70710678118654752f));
        if (RES) x += res[(size_t)(rbase + r) * N + colc];
        vals[r] = x;
      }
      if (OUTMODE == 0) {
        float* Cf = (float*)Cout;
        #pragma unroll
        for (int r = 0; r < 8; ++r) Cf[(size_t)(rbase + r) * N + colc] = vals[r];
      } else if (OUTMODE == 1) {
        unsigned short* Cb = (unsigned short*)Cout;
        #pragma unroll
        for (int r = 0; r < 8; ++r) Cb[(rbase + r) * N + colc] = f2bf(vals[r]);
      } else {
        // Vt[((b*H + h)*DV + dv)*S + s] : 8 consecutive tokens -> one uint4
        unsigned short* Cb = (unsigned short*)Cout;
        int bb = rbase >> 10;            // row / SS
        int sb = rbase & (SS - 1);
        int hh = colc >> 6;
        int dv = colc & 63;
        uint4 pk;
        pk.x = pk2bf(vals[0], vals[1]); pk.y = pk2bf(vals[2], vals[3]);
        pk.z = pk2bf(vals[4], vals[5]); pk.w = pk2bf(vals[6], vals[7]);
        *(uint4*)&Cb[((bb * HH + hh) * DVV + dv) * SS + sb] = pk;
      }
    }
  }
}

// ---------------------------------------------------------------------------
// Causal flash attention, one wave per (b, h, 16-query tile).
// Q,K bf16 [B,S,H*64]; V bf16 transposed [B,H,64,S]; O bf16 [B,S,H*64].
// 16-bit fragments are per-lane 8 consecutive K at half*8 (+16) -> uint4 loads.
// ---------------------------------------------------------------------------
__global__ __launch_bounds__(32) void attn_fa_wmma(
    const unsigned short* __restrict__ Q, const unsigned short* __restrict__ Kb,
    const unsigned short* __restrict__ Vt, unsigned short* __restrict__ O)
{
  __shared__ __align__(16) unsigned short Pl[16 * 32];  // P tile, bf16 row-major

  const int lane = threadIdx.x;
  const int half = lane >> 4;
  const int lm   = lane & 15;
  const int qt = blockIdx.x, h = blockIdx.y, b = blockIdx.z;
  const int qbase = qt * 16;

  // Q A-fragments (DK=64 -> 2 chunks of K=32); 32-bit offsets only
  Frag qf[2];
  const unsigned short* qrow = Q + (((b * SS + qbase + lm) * HH + h) * DKK);
  #pragma unroll
  for (int c = 0; c < 2; ++c) {
    *(uint4*)&qf[c].u[0] = *(const uint4*)(qrow + c * 32 + half * 8);
    *(uint4*)&qf[c].u[8] = *(const uint4*)(qrow + c * 32 + 16 + half * 8);
  }

  v8f oacc[4] = {{}, {}, {}, {}};
  float mst[8], lst[8];
  #pragma unroll
  for (int r = 0; r < 8; ++r) { mst[r] = -1e30f; lst[r] = 0.0f; }

  const int ntiles = (qbase + 15) / 32 + 1;   // causal: key tiles of 32
  for (int kt = 0; kt < ntiles; ++kt) {
    const int kb = kt * 32;
    v8f s0 = {}, s1 = {};
    #pragma unroll
    for (int t = 0; t < 2; ++t) {
      Frag kf0, kf1;
      const unsigned short* krow = Kb + (((b * SS + kb + t * 16 + lm) * HH + h) * DKK);
      *(uint4*)&kf0.u[0] = *(const uint4*)(krow + half * 8);
      *(uint4*)&kf0.u[8] = *(const uint4*)(krow + 16 + half * 8);
      *(uint4*)&kf1.u[0] = *(const uint4*)(krow + 32 + half * 8);
      *(uint4*)&kf1.u[8] = *(const uint4*)(krow + 48 + half * 8);
      if (t == 0) {
        s0 = __builtin_amdgcn_wmma_f32_16x16x32_bf16(false, qf[0].v, false, kf0.v,
                                                     (short)0, s0, false, false);
        s0 = __builtin_amdgcn_wmma_f32_16x16x32_bf16(false, qf[1].v, false, kf1.v,
                                                     (short)0, s0, false, false);
      } else {
        s1 = __builtin_amdgcn_wmma_f32_16x16x32_bf16(false, qf[0].v, false, kf0.v,
                                                     (short)0, s1, false, false);
        s1 = __builtin_amdgcn_wmma_f32_16x16x32_bf16(false, qf[1].v, false, kf1.v,
                                                     (short)0, s1, false, false);
      }
    }

    // scale + causal mask + online softmax (row spans 16 lanes of a half)
    #pragma unroll
    for (int r = 0; r < 8; ++r) {
      int row  = qbase + r + 8 * half;
      float v0 = (kb + lm      <= row) ? s0[r] * 0.125f : -1e30f;
      float v1 = (kb + 16 + lm <= row) ? s1[r] * 0.125f : -1e30f;
      float mt = fmaxf(v0, v1);
      #pragma unroll
      for (int msk = 1; msk < 16; msk <<= 1)
        mt = fmaxf(mt, __shfl_xor(mt, msk, 32));
      float mnew  = fmaxf(mst[r], mt);
      float alpha = __expf(mst[r] - mnew);
      float p0 = __expf(v0 - mnew);
      float p1 = __expf(v1 - mnew);
      float rs = p0 + p1;
      #pragma unroll
      for (int msk = 1; msk < 16; msk <<= 1)
        rs += __shfl_xor(rs, msk, 32);
      lst[r] = lst[r] * alpha + rs;
      mst[r] = mnew;
      oacc[0][r] *= alpha; oacc[1][r] *= alpha;
      oacc[2][r] *= alpha; oacc[3][r] *= alpha;
      s0[r] = p0; s1[r] = p1;
    }

    // P (C layout) -> LDS row-major -> A-fragment (two b128 per lane)
    __syncthreads();
    #pragma unroll
    for (int r = 0; r < 8; ++r) {
      int row = r + 8 * half;
      Pl[row * 32 + lm]      = f2bf(s0[r]);
      Pl[row * 32 + 16 + lm] = f2bf(s1[r]);
    }
    __syncthreads();
    Frag pf;
    *(uint4*)&pf.u[0] = *(const uint4*)&Pl[lm * 32 + half * 8];
    *(uint4*)&pf.u[8] = *(const uint4*)&Pl[lm * 32 + 16 + half * 8];

    // O += P @ V  (Vt rows contiguous in s -> two b128 per fragment)
    #pragma unroll
    for (int j = 0; j < 4; ++j) {
      Frag vf;
      const unsigned short* vrow = Vt + (((b * HH + h) * DVV + j * 16 + lm) * SS + kb);
      *(uint4*)&vf.u[0] = *(const uint4*)(vrow + half * 8);
      *(uint4*)&vf.u[8] = *(const uint4*)(vrow + 16 + half * 8);
      oacc[j] = __builtin_amdgcn_wmma_f32_16x16x32_bf16(false, pf.v, false, vf.v,
                                                        (short)0, oacc[j], false, false);
    }
  }

  // normalize + store (bf16)
  #pragma unroll
  for (int r = 0; r < 8; ++r) {
    int row = qbase + r + 8 * half;
    float inv = 1.0f / lst[r];
    #pragma unroll
    for (int j = 0; j < 4; ++j)
      O[((row + b * SS) * HH + h) * DVV + j * 16 + lm] = f2bf(oacc[j][r] * inv);
  }
}

// ---------------------------------------------------------------------------
// Host orchestration
// ---------------------------------------------------------------------------
extern "C" void kernel_launch(void* const* d_in, const int* in_sizes, int n_in,
                              void* d_out, int out_size, void* d_ws, size_t ws_size,
                              hipStream_t stream)
{
  const int*   tokens = (const int*)  d_in[0];
  const float* emb    = (const float*)d_in[1];
  const float* pos    = (const float*)d_in[2];
  const float* Wq     = (const float*)d_in[3];
  const float* bq     = (const float*)d_in[4];
  const float* Wk     = (const float*)d_in[5];
  const float* bk     = (const float*)d_in[6];
  const float* Wv     = (const float*)d_in[7];
  const float* bv     = (const float*)d_in[8];
  const float* Wo     = (const float*)d_in[9];
  const float* W1     = (const float*)d_in[10];
  const float* b1     = (const float*)d_in[11];
  const float* W2     = (const float*)d_in[12];
  const float* b2     = (const float*)d_in[13];
  const float* Wr     = (const float*)d_in[14];
  const float* br     = (const float*)d_in[15];

  const int M = BB * SS;                 // 2048
  const size_t tok_d = (size_t)M * DD;   // 2M elements

  char* ws = (char*)d_ws;
  float*          X   = (float*)ws;          ws += tok_d * 4;      // [M,D] f32
  float*          HB  = (float*)ws;          ws += tok_d * 4;      // post-attn resid f32
  unsigned short* Qb  = (unsigned short*)ws; ws += tok_d * 2;      // bf16 [B,S,H*DK]
  unsigned short* Kb  = (unsigned short*)ws; ws += tok_d * 2;      // bf16 [B,S,H*DK]
  unsigned short* Vb  = (unsigned short*)ws; ws += tok_d * 2;      // bf16 [B,H,DV,S]
  unsigned short* AO  = (unsigned short*)ws; ws += tok_d * 2;      // attn out bf16
  unsigned short* MH  = (unsigned short*)ws; ws += (size_t)M * DHH * 2; // [M,DH] bf16
  // per-layer swizzled bf16 weights (reused each layer)
  unsigned short* PWq = (unsigned short*)ws; ws += (size_t)DD * DD * 2;
  unsigned short* PWk = (unsigned short*)ws; ws += (size_t)DD * DD * 2;
  unsigned short* PWv = (unsigned short*)ws; ws += (size_t)DD * DD * 2;
  unsigned short* PWo = (unsigned short*)ws; ws += (size_t)DD * DD * 2;
  unsigned short* PW1 = (unsigned short*)ws; ws += (size_t)DD * DHH * 2;
  unsigned short* PW2 = (unsigned short*)ws; ws += (size_t)DHH * DD * 2;
  unsigned short* PWr = (unsigned short*)ws;                       // [D,V]

  embed_kernel<<<(unsigned)(tok_d / 4 / 256), 256, 0, stream>>>(tokens, emb, pos, X);
  // LM head weights converted once
  wconv_kernel<<<(unsigned)((size_t)DD * VV / 8 / 256), 256, 0, stream>>>(Wr, PWr, DD, VV);

  for (int l = 0; l < LL; ++l) {
    const float* wq = Wq + (size_t)l * DD * (HH * DKK);
    const float* wk = Wk + (size_t)l * DD * (HH * DKK);
    const float* wv = Wv + (size_t)l * DD * (HH * DVV);
    const float* wo = Wo + (size_t)l * (HH * DVV) * DD;
    const float* w1 = W1 + (size_t)l * DD * DHH;
    const float* w2 = W2 + (size_t)l * DHH * DD;
    const float* bql = bq + (size_t)l * (HH * DKK);
    const float* bkl = bk + (size_t)l * (HH * DKK);
    const float* bvl = bv + (size_t)l * (HH * DVV);
    const float* b1l = b1 + (size_t)l * DHH;
    const float* b2l = b2 + (size_t)l * DD;

    // convert+swizzle this layer's weights to bf16 (hot in L2 for the GEMMs)
    unsigned nb1 = (unsigned)((size_t)DD * DD / 8 / 256);
    unsigned nb4 = (unsigned)((size_t)DD * DHH / 8 / 256);
    wconv_kernel<<<nb1, 256, 0, stream>>>(wq, PWq, DD, DD);
    wconv_kernel<<<nb1, 256, 0, stream>>>(wk, PWk, DD, DD);
    wconv_kernel<<<nb1, 256, 0, stream>>>(wv, PWv, DD, DD);
    wconv_kernel<<<nb1, 256, 0, stream>>>(wo, PWo, DD, DD);
    wconv_kernel<<<nb4, 256, 0, stream>>>(w1, PW1, DD, DHH);
    wconv_kernel<<<nb4, 256, 0, stream>>>(w2, PW2, DHH, DD);

    dim3 gQKV((HH * DKK) / 128, M / 64);
    gemm_bf16_wmma<false, true,  false, false, 1><<<gQKV, 256, 0, stream>>>(X, PWq, bql, nullptr, Qb, M, HH * DKK, DD);
    gemm_bf16_wmma<false, true,  false, false, 1><<<gQKV, 256, 0, stream>>>(X, PWk, bkl, nullptr, Kb, M, HH * DKK, DD);
    gemm_bf16_wmma<false, true,  false, false, 2><<<gQKV, 256, 0, stream>>>(X, PWv, bvl, nullptr, Vb, M, HH * DVV, DD);

    attn_fa_wmma<<<dim3(SS / 16, HH, BB), 32, 0, stream>>>(Qb, Kb, Vb, AO);

    // h = x + o @ Wo        (A = bf16 attention output)
    gemm_bf16_wmma<true,  false, false, true,  0><<<dim3(DD / 128, M / 64), 256, 0, stream>>>(AO, PWo, nullptr, X, HB, M, DD, HH * DVV);
    // mlp hidden = gelu(h @ W1 + b1)   -> bf16
    gemm_bf16_wmma<false, true,  true,  false, 1><<<dim3(DHH / 128, M / 64), 256, 0, stream>>>(HB, PW1, b1l, nullptr, MH, M, DHH, DD);
    // x = h + (hidden @ W2 + b2)       (A = bf16 hidden)
    gemm_bf16_wmma<true,  true,  false, true,  0><<<dim3(DD / 128, M / 64), 256, 0, stream>>>(MH, PW2, b2l, HB, X, M, DD, DHH);
  }

  // logits = x @ Wr + br
  gemm_bf16_wmma<false, true, false, false, 0><<<dim3(VV / 128, M / 64), 256, 0, stream>>>(X, PWr, br, nullptr, (float*)d_out, M, VV, DD);
}
